// UserSocialGraph_1821066134232
// MI455X (gfx1250) — matrix operand order repaired and checked by reading, not compiled, not run on hardware
//
#include <hip/hip_runtime.h>
#include <hip/hip_bf16.h>

// ---------------------------------------------------------------------------
// GraphRec-style forward pass for MI455X (gfx1250, wave32, WMMA).
// All matmuls use v_wmma_f32_16x16x32_f16 (f16 A/B, f32 accumulate).
// ---------------------------------------------------------------------------

typedef __attribute__((ext_vector_type(16))) _Float16 v16h;
typedef __attribute__((ext_vector_type(8)))  float    v8f;

#define DD     128      // feature dim
#define NROWS  8192     // 2*B user rows
#define BB     4096     // batch
#define HH     50       // history length
#define HPAD   64       // padded history rows (4 M-tiles)
#define SS     32       // social neighbors (2 M-tiles)

// A-fragment (16x32 f16, M x K) per-lane K offset for element i (0..15):
//   lanes 0-15 : M = lane, K = {0..7, 16..23}
//   lanes 16-31: M = lane-16, K = {8..15, 24..31}
__device__ __forceinline__ int a_koff(int i, int hi) {
  return ((i >> 3) << 4) + (hi << 3) + (i & 7);
}

__device__ __forceinline__ v8f wmma_f16(v16h a, v16h b, v8f c) {
  // D = A(16x32) * B(32x16) + C, f32 accumulate
  return __builtin_amdgcn_wmma_f32_16x16x32_f16(
      /*neg_a=*/false, a, /*neg_b=*/false, b,
      /*c_mod=*/(short)0, c, /*reuse_a=*/false, /*reuse_b=*/false);
}

// ---------------------------------------------------------------------------
// Pack a (K x 128) f32 weight matrix into WMMA-B fragment order:
// half index = (((kt*8 + nt)*32) + lane)*16 + i, with
//   K = kt*32 + (i>>3)*16 + (lane>>4)*8 + (i&7),  N = nt*16 + (lane&15)
// so each lane's 16-half B fragment is one contiguous 32-byte load.
// ---------------------------------------------------------------------------
__global__ void pack_b_kernel(const float* __restrict__ W,
                              _Float16* __restrict__ P, int K) {
  int idx = blockIdx.x * blockDim.x + threadIdx.x;
  if (idx >= K * DD) return;
  int i    = idx & 15;
  int lane = (idx >> 4) & 31;
  int t    = idx >> 9;
  int nt   = t & 7;
  int kt   = t >> 3;
  int k = kt * 32 + ((i >> 3) << 4) + ((lane >> 4) << 3) + (i & 7);
  int n = nt * 16 + (lane & 15);
  P[idx] = (_Float16)W[k * DD + n];
}

// ---------------------------------------------------------------------------
// History branch: one workgroup (8 waves) per user row n.
//   X1 = relu([v2e|r2e](64x256) @ gv_w1 + b1)
//   X2 = relu(X1 @ gv_w2 + b2)
//   e  = relu([X2|u](64x256) @ atth_w1 + b1) . atth_w2   (per-row logit)
//   a  = softmax(e over 50), agg = sum a*X2 ; write UA = [u | agg]
// ---------------------------------------------------------------------------
__global__ __launch_bounds__(256) void hist_kernel(
    const float* __restrict__ u2e, const float* __restrict__ v2e,
    const float* __restrict__ r2e,
    const v16h* __restrict__ gv_w1p, const float* __restrict__ gv_b1,
    const v16h* __restrict__ gv_w2p, const float* __restrict__ gv_b2,
    const v16h* __restrict__ atth_w1p, const float* __restrict__ atth_b1,
    const float* __restrict__ atth_w2,
    const int* __restrict__ user_ids, const int* __restrict__ hist_items,
    const int* __restrict__ hist_ratings, float* __restrict__ UA) {
  __shared__ float    u_s[DD];
  __shared__ _Float16 u_h[DD];
  __shared__ _Float16 X1h[HPAD * DD];   // 16 KB
  __shared__ float    X2f[HPAD * DD];   // 32 KB
  __shared__ float    e_s[HPAD];
  __shared__ float    p_s[HPAD];
  __shared__ int      it_s[HPAD];
  __shared__ int      rt_s[HPAD];
  __shared__ float    red[2];

  const int n    = blockIdx.x;
  const int tid  = threadIdx.x;
  const int lane = tid & 31;
  const int wv   = tid >> 5;
  const int hi   = lane >> 4;
  const int lm   = lane & 15;

  if (tid < DD) {
    float uv = u2e[(long)user_ids[n] * DD + tid];
    u_s[tid] = uv;
    u_h[tid] = (_Float16)uv;
  }
  if (tid < HPAD) {
    int hh = tid < HH ? tid : (HH - 1);   // clamp pad rows (avoid OOB gather)
    it_s[tid] = hist_items[n * HH + hh];
    rt_s[tid] = hist_ratings[n * HH + hh];
    e_s[tid]  = 0.0f;
  }
  __syncthreads();

  // ---- GEMM1: (64x256) @ gv_w1 -> X1h (f16), K = 256 -> 8 k-tiles ----
  for (int t = wv; t < 32; t += 8) {
    int mt = t >> 3, nt = t & 7;
    int m  = (mt << 4) + lm;
    const float* vrow = v2e + (long)it_s[m] * DD;
    const float* rrow = r2e + (long)rt_s[m] * DD;
    v8f acc = {};
    for (int kt = 0; kt < 8; ++kt) {
      v16h a;
#pragma unroll
      for (int i = 0; i < 16; ++i) {
        int kk = kt * 32 + a_koff(i, hi);
        float v = (kk < DD) ? vrow[kk] : rrow[kk - DD];
        a[i] = (_Float16)v;
      }
      v16h b = gv_w1p[(kt * 8 + nt) * 32 + lane];
      acc = wmma_f16(a, b, acc);
    }
    int col = (nt << 4) + lm;
    float bv = gv_b1[col];
#pragma unroll
    for (int r = 0; r < 8; ++r) {
      int row = (mt << 4) + (hi << 3) + r;
      X1h[row * DD + col] = (_Float16)fmaxf(acc[r] + bv, 0.0f);
    }
  }
  __syncthreads();

  // ---- GEMM2: X1 @ gv_w2 -> X2f (f32), K = 128 -> 4 k-tiles ----
  for (int t = wv; t < 32; t += 8) {
    int mt = t >> 3, nt = t & 7;
    int m  = (mt << 4) + lm;
    v8f acc = {};
    for (int kt = 0; kt < 4; ++kt) {
      v16h a;
#pragma unroll
      for (int i = 0; i < 16; ++i)
        a[i] = X1h[m * DD + kt * 32 + a_koff(i, hi)];
      v16h b = gv_w2p[(kt * 8 + nt) * 32 + lane];
      acc = wmma_f16(a, b, acc);
    }
    int col = (nt << 4) + lm;
    float bv = gv_b2[col];
#pragma unroll
    for (int r = 0; r < 8; ++r) {
      int row = (mt << 4) + (hi << 3) + r;
      X2f[row * DD + col] = fmaxf(acc[r] + bv, 0.0f);
    }
  }
  __syncthreads();

  // ---- GEMM3: [X2|u] @ atth_w1, fused dot with atth_w2 into e_s ----
  for (int t = wv; t < 32; t += 8) {
    int mt = t >> 3, nt = t & 7;
    int m  = (mt << 4) + lm;
    v8f acc = {};
    for (int kt = 0; kt < 8; ++kt) {
      v16h a;
#pragma unroll
      for (int i = 0; i < 16; ++i) {
        int kk = kt * 32 + a_koff(i, hi);
        a[i] = (kk < DD) ? (_Float16)X2f[m * DD + kk] : u_h[kk - DD];
      }
      v16h b = atth_w1p[(kt * 8 + nt) * 32 + lane];
      acc = wmma_f16(a, b, acc);
    }
    int col = (nt << 4) + lm;
    float bv = atth_b1[col];
    float w2 = atth_w2[col];
#pragma unroll
    for (int r = 0; r < 8; ++r) {
      int row = (mt << 4) + (hi << 3) + r;
      float v = fmaxf(acc[r] + bv, 0.0f);
      atomicAdd(&e_s[row], v * w2);   // ds_add_f32
    }
  }
  __syncthreads();

  // ---- softmax over 50 + weighted aggregate (atth_b2 drops out) ----
  if (tid == 0) {
    float mx = -3.0e38f;
    for (int h = 0; h < HH; ++h) mx = fmaxf(mx, e_s[h]);
    red[0] = mx;
  }
  __syncthreads();
  if (tid < HH) p_s[tid] = __expf(e_s[tid] - red[0]);
  __syncthreads();
  if (tid == 0) {
    float s = 0.0f;
    for (int h = 0; h < HH; ++h) s += p_s[h];
    red[1] = s;
  }
  __syncthreads();
  if (tid < DD) {
    float inv = 1.0f / red[1];
    float s = 0.0f;
    for (int h = 0; h < HH; ++h) s += p_s[h] * X2f[h * DD + tid];
    UA[(long)n * 2 * DD + tid]      = u_s[tid];       // cols   0..127 : u
    UA[(long)n * 2 * DD + DD + tid] = s * inv;        // cols 128..255 : agg
  }
}

// ---------------------------------------------------------------------------
// Social branch: 32 neighbors per row, same pattern (u2e L2-resident).
// Writes agg_s into SA cols 128..255 (cols 0..127 filled by enc GEMM).
// ---------------------------------------------------------------------------
__global__ __launch_bounds__(256) void soc_kernel(
    const float* __restrict__ u2e,
    const v16h* __restrict__ atts_w1p, const float* __restrict__ atts_b1,
    const float* __restrict__ atts_w2,
    const int* __restrict__ user_ids, const int* __restrict__ soc_neigh,
    float* __restrict__ SA) {
  __shared__ float    u_s[DD];
  __shared__ _Float16 u_h[DD];
  __shared__ int      nb_s[SS];
  __shared__ float    e_s[SS];
  __shared__ float    p_s[SS];
  __shared__ float    red[2];

  const int n    = blockIdx.x;
  const int tid  = threadIdx.x;
  const int lane = tid & 31;
  const int wv   = tid >> 5;
  const int hi   = lane >> 4;
  const int lm   = lane & 15;

  if (tid < DD) {
    float uv = u2e[(long)user_ids[n] * DD + tid];
    u_s[tid] = uv;
    u_h[tid] = (_Float16)uv;
  }
  if (tid < SS) {
    nb_s[tid] = soc_neigh[n * SS + tid];
    e_s[tid]  = 0.0f;
  }
  __syncthreads();

  for (int t = wv; t < 16; t += 8) {
    int mt = t >> 3, nt = t & 7;
    int m  = (mt << 4) + lm;
    const float* nrow = u2e + (long)nb_s[m] * DD;
    v8f acc = {};
    for (int kt = 0; kt < 8; ++kt) {
      v16h a;
#pragma unroll
      for (int i = 0; i < 16; ++i) {
        int kk = kt * 32 + a_koff(i, hi);
        a[i] = (kk < DD) ? (_Float16)nrow[kk] : u_h[kk - DD];
      }
      v16h b = atts_w1p[(kt * 8 + nt) * 32 + lane];
      acc = wmma_f16(a, b, acc);
    }
    int col = (nt << 4) + lm;
    float bv = atts_b1[col];
    float w2 = atts_w2[col];
#pragma unroll
    for (int r = 0; r < 8; ++r) {
      int row = (mt << 4) + (hi << 3) + r;
      float v = fmaxf(acc[r] + bv, 0.0f);
      atomicAdd(&e_s[row], v * w2);
    }
  }
  __syncthreads();

  if (tid == 0) {
    float mx = -3.0e38f;
    for (int s2 = 0; s2 < SS; ++s2) mx = fmaxf(mx, e_s[s2]);
    red[0] = mx;
  }
  __syncthreads();
  if (tid < SS) p_s[tid] = __expf(e_s[tid] - red[0]);
  __syncthreads();
  if (tid == 0) {
    float s = 0.0f;
    for (int s2 = 0; s2 < SS; ++s2) s += p_s[s2];
    red[1] = s;
  }
  __syncthreads();
  if (tid < DD) {
    float inv = 1.0f / red[1];
    float s = 0.0f;
    for (int s2 = 0; s2 < SS; ++s2)
      s += p_s[s2] * u2e[(long)nb_s[s2] * DD + tid];
    SA[(long)n * 2 * DD + DD + tid] = s * inv;
  }
}

// ---------------------------------------------------------------------------
// Generic WMMA GEMM: C[M x 128] = act(A[M x K](f32) @ Wpacked + bias)
// One wave per 16x16 output tile; 8 waves per block. act: 0=none,1=relu,2=tanh
// ---------------------------------------------------------------------------
__global__ __launch_bounds__(256) void gemm_kernel(
    const float* __restrict__ A, int lda,
    const v16h* __restrict__ Wp, const float* __restrict__ bias,
    float* __restrict__ C, int ldc, int M, int K, int act) {
  const int lane = threadIdx.x & 31;
  const int wv   = threadIdx.x >> 5;
  const int hi   = lane >> 4;
  const int lm   = lane & 15;
  const int tiles = (M >> 4) * 8;
  int tile = blockIdx.x * 8 + wv;
  if (tile >= tiles) return;
  int mt = tile >> 3, nt = tile & 7;
  int m  = (mt << 4) + lm;
  const float* Arow = A + (long)m * lda;
  v8f acc = {};
  const int kts = K >> 5;
  for (int kt = 0; kt < kts; ++kt) {
    v16h a;
#pragma unroll
    for (int i = 0; i < 16; ++i)
      a[i] = (_Float16)Arow[kt * 32 + a_koff(i, hi)];
    v16h b = Wp[(kt * 8 + nt) * 32 + lane];
    acc = wmma_f16(a, b, acc);
  }
  int col  = (nt << 4) + lm;
  float bv = bias ? bias[col] : 0.0f;
#pragma unroll
  for (int r = 0; r < 8; ++r) {
    int row = (mt << 4) + (hi << 3) + r;
    float v = acc[r] + bv;
    if (act == 1)      v = fmaxf(v, 0.0f);
    else if (act == 2) v = tanhf(v);
    C[(long)row * ldc + col] = v;
  }
}

// ---- pair attention: ap = softmax over 2 of tanh(h@Wa)@ba ; Wq = ap*hp ----
__global__ __launch_bounds__(128) void pair_attn_kernel(
    const float* __restrict__ T, const float* __restrict__ ba,
    const float* __restrict__ Hm, float* __restrict__ Wq) {
  __shared__ float r0[DD], r1[DD];
  __shared__ float a01[2];
  int b = blockIdx.x, tid = threadIdx.x;
  long i0 = (long)(2 * b) * DD + tid, i1 = i0 + DD;
  r0[tid] = T[i0] * ba[tid];
  r1[tid] = T[i1] * ba[tid];
  __syncthreads();
  for (int s = 64; s > 0; s >>= 1) {
    if (tid < s) { r0[tid] += r0[tid + s]; r1[tid] += r1[tid + s]; }
    __syncthreads();
  }
  if (tid == 0) {
    float t0 = r0[0], t1 = r1[0];
    float mx = fmaxf(t0, t1);
    float e0 = __expf(t0 - mx), e1 = __expf(t1 - mx);
    float inv = 1.0f / (e0 + e1);
    a01[0] = e0 * inv; a01[1] = e1 * inv;
  }
  __syncthreads();
  Wq[i0] = a01[0] * Hm[i0];
  Wq[i1] = a01[1] * Hm[i1];
}

// ---- normalization across the pair axis (axis=1 of (B,2,D)) + relu ----
__global__ void pairnorm_kernel(const float* __restrict__ Y,
                                const float* __restrict__ g,
                                const float* __restrict__ bb,
                                float* __restrict__ O, int Bn) {
  int idx = blockIdx.x * blockDim.x + threadIdx.x;
  if (idx >= Bn * DD) return;
  int b = idx >> 7, d = idx & 127;
  long i0 = (long)(2 * b) * DD + d, i1 = i0 + DD;
  float y0 = Y[i0], y1 = Y[i1];
  float mu = 0.5f * (y0 + y1);
  float d0 = y0 - mu, d1 = y1 - mu;
  float inv = rsqrtf(0.5f * (d0 * d0 + d1 * d1) + 1e-5f);
  float gg = g[d], bv = bb[d];
  O[i0] = fmaxf(d0 * inv * gg + bv, 0.0f);
  O[i1] = fmaxf(d1 * inv * gg + bv, 0.0f);
}

// ---- ini = y[:,0], res = y[:,1] ----
__global__ void extract_kernel(const float* __restrict__ Y,
                               float* __restrict__ ini,
                               float* __restrict__ res, int Bn) {
  int idx = blockIdx.x * blockDim.x + threadIdx.x;
  if (idx >= Bn * DD) return;
  int b = idx >> 7, d = idx & 127;
  ini[idx] = Y[(long)(2 * b) * DD + d];
  res[idx] = Y[(long)(2 * b + 1) * DD + d];
}

// ---- per-column mean/var over batch (for batchnorm over axis 0) ----
__global__ __launch_bounds__(256) void colstats_kernel(
    const float* __restrict__ Z, float* __restrict__ st, int Mr) {
  __shared__ float ss[256], sq[256];
  int d = blockIdx.x, tid = threadIdx.x;
  float s = 0.0f, q = 0.0f;
  for (int r = tid; r < Mr; r += 256) {
    float v = Z[(long)r * DD + d];
    s += v; q += v * v;
  }
  ss[tid] = s; sq[tid] = q;
  __syncthreads();
  for (int st2 = 128; st2 > 0; st2 >>= 1) {
    if (tid < st2) { ss[tid] += ss[tid + st2]; sq[tid] += sq[tid + st2]; }
    __syncthreads();
  }
  if (tid == 0) {
    float mean = ss[0] / (float)Mr;
    st[d]      = mean;
    st[DD + d] = sq[0] / (float)Mr - mean * mean;
  }
}

__global__ void bnorm_kernel(const float* __restrict__ Z,
                             const float* __restrict__ st,
                             const float* __restrict__ g,
                             const float* __restrict__ bb,
                             float* __restrict__ O, int Bn) {
  int idx = blockIdx.x * blockDim.x + threadIdx.x;
  if (idx >= Bn * DD) return;
  int d = idx & 127;
  float v = (Z[idx] - st[d]) * rsqrtf(st[DD + d] + 1e-5f) * g[d] + bb[d];
  O[idx] = fmaxf(v, 0.0f);
}

// ---------------------------------------------------------------------------
extern "C" void kernel_launch(void* const* d_in, const int* in_sizes, int n_in,
                              void* d_out, int out_size, void* d_ws,
                              size_t ws_size, hipStream_t stream) {
  (void)in_sizes; (void)n_in; (void)out_size; (void)ws_size;

  const float* u2e      = (const float*)d_in[0];
  const float* v2e      = (const float*)d_in[1];
  const float* r2e      = (const float*)d_in[2];
  const float* gv_w1    = (const float*)d_in[3];
  const float* gv_b1    = (const float*)d_in[4];
  const float* gv_w2    = (const float*)d_in[5];
  const float* gv_b2    = (const float*)d_in[6];
  const float* atth_w1  = (const float*)d_in[7];
  const float* atth_b1  = (const float*)d_in[8];
  const float* atth_w2  = (const float*)d_in[9];
  // d_in[10] atth_b2: constant shift, softmax-invariant -> skipped
  const float* enc_w    = (const float*)d_in[11];
  const float* enc_b    = (const float*)d_in[12];
  const float* atts_w1  = (const float*)d_in[13];
  const float* atts_b1  = (const float*)d_in[14];
  const float* atts_w2  = (const float*)d_in[15];
  // d_in[16] atts_b2: softmax-invariant -> skipped
  const float* soc_w    = (const float*)d_in[17];
  const float* soc_b    = (const float*)d_in[18];
  const float* attn_wa  = (const float*)d_in[19];
  const float* attn_ba  = (const float*)d_in[20];
  const float* fc1_w    = (const float*)d_in[21];
  const float* fc1_b    = (const float*)d_in[22];
  const float* bn1_g    = (const float*)d_in[23];
  const float* bn1_b    = (const float*)d_in[24];
  const float* fc2_w    = (const float*)d_in[25];
  const float* fc2_b    = (const float*)d_in[26];
  const float* efc1_w   = (const float*)d_in[27];
  const float* efc1_b   = (const float*)d_in[28];
  const float* bn2_g    = (const float*)d_in[29];
  const float* bn2_b    = (const float*)d_in[30];
  const float* efc2_w   = (const float*)d_in[31];
  const float* efc2_b   = (const float*)d_in[32];
  const int* user_ids     = (const int*)d_in[33];
  const int* hist_items   = (const int*)d_in[34];
  const int* hist_ratings = (const int*)d_in[35];
  const int* soc_neigh    = (const int*)d_in[36];

  // ---- workspace layout (~21.5 MB peak, with aliased reuse) ----
  char* ws = (char*)d_ws;
  _Float16* gv_w1p   = (_Float16*)ws;          // 32768 halves each 256xK pack
  _Float16* gv_w2p   = gv_w1p   + 32768;       // 16384
  _Float16* atth_w1p = gv_w2p   + 16384;       // 32768
  _Float16* atts_w1p = atth_w1p + 32768;       // 32768
  _Float16* enc_wp   = atts_w1p + 32768;       // 32768
  _Float16* soc_wp   = enc_wp   + 32768;       // 32768
  _Float16* attn_wap = soc_wp   + 32768;       // 16384
  _Float16* fc1_wp   = attn_wap + 16384;       // 16384
  _Float16* fc2_wp   = fc1_wp   + 16384;       // 16384
  _Float16* efc1_wp  = fc2_wp   + 16384;       // 32768
  _Float16* efc2_wp  = efc1_wp  + 32768;       // 16384  (ends @ 557056 B)
  float* stats = (float*)(ws + 557056);        // 256 floats
  float* UA    = (float*)(ws + 0x90000);       // 8192 x 256
  float* SA    = UA + (long)NROWS * 2 * DD;    // 8192 x 256
  float* Hmat  = SA + (long)NROWS * 2 * DD;    // 8192 x 128
  float* T  = UA;                   // alias: UA dead after enc GEMM
  float* Wq = UA + (long)NROWS * DD;
  float* Y1 = SA;                   // alias: SA dead after soc GEMM
  float* Y3 = SA + (long)NROWS * DD;
  float* Y2 = Hmat;                 // alias: Hmat dead after pair_attn
  float* Z1 = UA;                   // alias: T dead after pair_attn
  float* Z2 = UA + (long)NROWS * DD;// alias: Wq dead after fc1
  float* out = (float*)d_out;       // [z(4096x128) | ini | res]

  auto pack = [&](const float* w, _Float16* p, int K) {
    int n = K * DD;
    pack_b_kernel<<<(n + 255) / 256, 256, 0, stream>>>(w, p, K);
  };
  pack(gv_w1, gv_w1p, 256);   pack(gv_w2, gv_w2p, 128);
  pack(atth_w1, atth_w1p, 256); pack(atts_w1, atts_w1p, 256);
  pack(enc_w, enc_wp, 256);   pack(soc_w, soc_wp, 256);
  pack(attn_wa, attn_wap, 128); pack(fc1_w, fc1_wp, 128);
  pack(fc2_w, fc2_wp, 128);   pack(efc1_w, efc1_wp, 256);
  pack(efc2_w, efc2_wp, 128);

  hist_kernel<<<NROWS, 256, 0, stream>>>(
      u2e, v2e, r2e, (const v16h*)gv_w1p, gv_b1, (const v16h*)gv_w2p, gv_b2,
      (const v16h*)atth_w1p, atth_b1, atth_w2, user_ids, hist_items,
      hist_ratings, UA);
  soc_kernel<<<NROWS, 256, 0, stream>>>(
      u2e, (const v16h*)atts_w1p, atts_b1, atts_w2, user_ids, soc_neigh, SA);

  auto gemm = [&](const float* A, int lda, _Float16* p, const float* bias,
                  float* C, int ldc, int M, int K, int act) {
    int tiles = (M >> 4) * 8;
    gemm_kernel<<<(tiles + 7) / 8, 256, 0, stream>>>(
        A, lda, (const v16h*)p, bias, C, ldc, M, K, act);
  };

  gemm(UA, 256, enc_wp, enc_b, SA, 256, NROWS, 256, 1);    // h_hist -> SA[:,0:128]
  gemm(SA, 256, soc_wp, soc_b, Hmat, 128, NROWS, 256, 1);  // h
  gemm(Hmat, 128, attn_wap, nullptr, T, 128, NROWS, 128, 2); // tanh(h@Wa)
  pair_attn_kernel<<<BB, 128, 0, stream>>>(T, attn_ba, Hmat, Wq);
  gemm(Wq, 128, fc1_wp, fc1_b, Y1, 128, NROWS, 128, 0);    // w@fc1 + b
  pairnorm_kernel<<<(BB * DD + 255) / 256, 256, 0, stream>>>(Y1, bn1_g, bn1_b,
                                                             Y2, BB);
  gemm(Y2, 128, fc2_wp, fc2_b, Y3, 128, NROWS, 128, 0);    // y
  extract_kernel<<<(BB * DD + 255) / 256, 256, 0, stream>>>(
      Y3, out + (long)BB * DD, out + 2L * BB * DD, BB);    // ini, res
  gemm(Y3, 256, efc1_wp, efc1_b, Z1, 128, BB, 256, 0);     // c@efc1
  colstats_kernel<<<DD, 256, 0, stream>>>(Z1, stats, BB);
  bnorm_kernel<<<(BB * DD + 255) / 256, 256, 0, stream>>>(Z1, stats, bn2_g,
                                                          bn2_b, Z2, BB);
  gemm(Z2, 128, efc2_wp, efc2_b, out, 128, BB, 128, 0);    // z -> d_out
}
